// LearnableCurvGCN_88356067213587
// MI455X (gfx1250) — compile-verified, matrix operand order checked by reading, not run on hardware
//
#include <hip/hip_runtime.h>
#include <math.h>

// ---------------------------------------------------------------------------
// CDNA5 (gfx1250) types for WMMA
// ---------------------------------------------------------------------------
typedef __attribute__((ext_vector_type(16))) __bf16 v16bf;
typedef __attribute__((ext_vector_type(8)))  float  v8f;

#define N_NODES 100000
#define N_EDGES 1600000
#define IN_DIM  128
#define HID_DIM 128
#define OUT_DIM 64

// ---------------------------------------------------------------------------
// Utility kernels
// ---------------------------------------------------------------------------
__global__ void zero_f32_kernel(float* __restrict__ p, int n) {
    int i = blockIdx.x * blockDim.x + threadIdx.x;
    if (i < n) p[i] = 0.0f;
}

__global__ void relu_inplace_kernel(float* __restrict__ p, int n) {
    int i = blockIdx.x * blockDim.x + threadIdx.x;
    if (i < n) p[i] = fmaxf(p[i], 0.0f);
}

// dis = rsqrt(deg + 1)   (deg accumulated from positive edge weights, so > 0)
__global__ void rsqrt_deg_kernel(float* __restrict__ deg, int n) {
    int i = blockIdx.x * blockDim.x + threadIdx.x;
    if (i < n) deg[i] = rsqrtf(deg[i] + 1.0f);
}

// ---------------------------------------------------------------------------
// Fused per-edge curvature MLP:  Linear(1,16)->ReLU->Linear(16,1)->Sigmoid
// edge_weight = 0.1 + 0.9*sigmoid(...),   deg[dst] += edge_weight
// ---------------------------------------------------------------------------
__global__ void edge_mlp_kernel(const float* __restrict__ curv,
                                const int*   __restrict__ dst,
                                const float* __restrict__ mw1,
                                const float* __restrict__ mb1,
                                const float* __restrict__ mw2,
                                const float* __restrict__ mb2,
                                float* __restrict__ ew,
                                float* __restrict__ deg,
                                int nE) {
    int e = blockIdx.x * blockDim.x + threadIdx.x;
    if (e >= nE) return;
    float c = curv[e];
    float s = mb2[0];
#pragma unroll
    for (int j = 0; j < 16; ++j) {
        float h = fmaf(c, mw1[j], mb1[j]);
        h = fmaxf(h, 0.0f);
        s = fmaf(h, mw2[j], s);
    }
    float sg = 1.0f / (1.0f + __expf(-s));
    float w  = fmaf(0.9f, sg, 0.1f);
    ew[e] = w;
    atomicAdd(&deg[dst[e]], w);
}

// ---------------------------------------------------------------------------
// WMMA GEMM:  XW = X @ W     X:[M,128] f32, W:[128,NDIM] f32, XW:[M,NDIM] f32
// One wave per 16x16 output tile.  K=128 -> 4x v_wmma_f32_16x16x32_bf16.
// Register layouts follow CDNA5 ISA 7.12.2 (16-bit A 16x32; B mirrored on N;
// 32-bit C/D 16x16).  M must be a multiple of 16 (100000 = 6250*16).
// ---------------------------------------------------------------------------
template <int NDIM>
__global__ void __launch_bounds__(256)
gemm_wmma_kernel(const float* __restrict__ X,
                 const float* __restrict__ W,
                 float* __restrict__ XW,
                 int mtiles) {
    constexpr int NT = NDIM / 16;                 // N tiles
    const int wave = threadIdx.x >> 5;
    const int lane = threadIdx.x & 31;
    const int t = blockIdx.x * 8 + wave;          // flat tile id, whole-wave
    if (t >= mtiles * NT) return;                 // full-wave exit: EXEC stays all-1s
    const int tm = t / NT;
    const int tn = t % NT;

    const int half = lane >> 4;                   // K-group selector
    const int idx  = lane & 15;                   // A row / B col within tile
    const size_t arow = (size_t)(tm * 16 + idx) * 128;
    const int    col  = tn * 16 + idx;

    v8f acc = {};
#pragma unroll
    for (int ks = 0; ks < 128; ks += 32) {
        v16bf a, b;
#pragma unroll
        for (int v = 0; v < 8; ++v) {
            // ISA 16-bit A layout: VGPR v<4 -> K = half*8 + 2v (+1),
            //                      VGPR v>=4 -> +16.  All K offsets even.
            const int koff = ((v & 4) ? 16 : 0) + half * 8 + 2 * (v & 3);
            const int k = ks + koff;
            const float2 f = *(const float2*)(X + arow + k);
            a[2 * v]     = (__bf16)f.x;
            a[2 * v + 1] = (__bf16)f.y;
            b[2 * v]     = (__bf16)W[(size_t)k * NDIM + col];
            b[2 * v + 1] = (__bf16)W[(size_t)(k + 1) * NDIM + col];
        }
        acc = __builtin_amdgcn_wmma_f32_16x16x32_bf16(
            /*neg_a=*/false, a, /*neg_b=*/false, b,
            /*c_mod=*/(short)0, acc, /*reuse_a=*/false, /*reuse_b=*/false);
    }
    // C/D layout: VGPR r, lanes 0-15: M=r; lanes 16-31: M=8+r; N=lane&15
#pragma unroll
    for (int r = 0; r < 8; ++r) {
        XW[(size_t)(tm * 16 + half * 8 + r) * NDIM + tn * 16 + idx] = acc[r];
    }
}

// ---------------------------------------------------------------------------
// out[i,d] = dis[i]^2 * xw[i,d] + b[d]        (self-loop term + bias seed)
// ---------------------------------------------------------------------------
__global__ void self_bias_kernel(const float* __restrict__ xw,
                                 const float* __restrict__ dis,
                                 const float* __restrict__ b,
                                 float* __restrict__ out,
                                 int total, int D) {
    int i = blockIdx.x * blockDim.x + threadIdx.x;
    if (i >= total) return;
    int n = i / D;
    int d = i - n * D;
    float di = dis[n];
    out[i] = fmaf(di * di, xw[i], b[d]);
}

// ---------------------------------------------------------------------------
// Edge scatter:  out[dst] += dis[src]*ew*dis[dst] * xw[src]
// One wave per edge; each lane handles VEC consecutive features.
// ---------------------------------------------------------------------------
template <int VEC>   // VEC = D/32
__global__ void __launch_bounds__(256)
scatter_kernel(const int* __restrict__ src,
               const int* __restrict__ dst,
               const float* __restrict__ ew,
               const float* __restrict__ dis,
               const float* __restrict__ xw,
               float* __restrict__ out,
               int nE) {
    constexpr int D = VEC * 32;
    const int wid  = (int)((blockIdx.x * blockDim.x + threadIdx.x) >> 5);
    const int lane = threadIdx.x & 31;
    if (wid >= nE) return;
    const int s = src[wid];
    const int d = dst[wid];
    const float nrm = dis[s] * ew[wid] * dis[d];
    const float* xr = xw + (size_t)s * D + lane * VEC;
    float*       orow = out + (size_t)d * D + lane * VEC;
#pragma unroll
    for (int i = 0; i < VEC; ++i) {
        atomicAdd(&orow[i], nrm * xr[i]);
    }
}

// ---------------------------------------------------------------------------
// Launcher
// ---------------------------------------------------------------------------
static inline size_t align256(size_t x) { return (x + 255) & ~(size_t)255; }

extern "C" void kernel_launch(void* const* d_in, const int* in_sizes, int n_in,
                              void* d_out, int out_size, void* d_ws, size_t ws_size,
                              hipStream_t stream) {
    (void)in_sizes; (void)n_in; (void)out_size; (void)ws_size;
    // setup_inputs() order:
    const float* x    = (const float*)d_in[0];            // [N,128]
    const int*   ei   = (const int*)  d_in[1];            // [2,E]
    const float* curv = (const float*)d_in[2];            // [E,1]
    const float* W1   = (const float*)d_in[3];            // [128,128]
    const float* b1   = (const float*)d_in[4];            // [128]
    const float* W2   = (const float*)d_in[5];            // [128,64]
    const float* b2   = (const float*)d_in[6];            // [64]
    const float* mw1  = (const float*)d_in[7];            // [1,16]
    const float* mb1  = (const float*)d_in[8];            // [16]
    const float* mw2  = (const float*)d_in[9];            // [16,1]
    const float* mb2  = (const float*)d_in[10];           // [1]
    float* out = (float*)d_out;                           // [N,64]

    const int* srcp = ei;
    const int* dstp = ei + N_EDGES;

    // workspace layout
    char* ws = (char*)d_ws;
    float* ew  = (float*)ws;  ws += align256((size_t)N_EDGES * 4);
    float* dis = (float*)ws;  ws += align256((size_t)N_NODES * 4);   // deg then dis
    float* xw1 = (float*)ws;  ws += align256((size_t)N_NODES * HID_DIM * 4);
    float* h1  = (float*)ws;  ws += align256((size_t)N_NODES * HID_DIM * 4);
    float* xw2 = (float*)ws;  ws += align256((size_t)N_NODES * OUT_DIM * 4);

    const int T = 256;

    // 1) deg = 0
    zero_f32_kernel<<<(N_NODES + T - 1) / T, T, 0, stream>>>(dis, N_NODES);
    // 2) edge MLP -> ew, deg[dst] += ew
    edge_mlp_kernel<<<(N_EDGES + T - 1) / T, T, 0, stream>>>(
        curv, dstp, mw1, mb1, mw2, mb2, ew, dis, N_EDGES);
    // 3) dis = rsqrt(deg + 1)
    rsqrt_deg_kernel<<<(N_NODES + T - 1) / T, T, 0, stream>>>(dis, N_NODES);

    const int mtiles = N_NODES / 16;                      // 6250
    // 4) xw1 = x @ W1   (WMMA bf16, f32 accumulate)
    {
        int tiles = mtiles * (HID_DIM / 16);              // 50000
        gemm_wmma_kernel<HID_DIM><<<(tiles + 7) / 8, 256, 0, stream>>>(x, W1, xw1, mtiles);
    }
    // 5) h1 = dis^2*xw1 + b1
    self_bias_kernel<<<(N_NODES * HID_DIM + T - 1) / T, T, 0, stream>>>(
        xw1, dis, b1, h1, N_NODES * HID_DIM, HID_DIM);
    // 6) h1[dst] += norm * xw1[src]
    scatter_kernel<HID_DIM / 32><<<(N_EDGES * 32 + T - 1) / T, T, 0, stream>>>(
        srcp, dstp, ew, dis, xw1, h1, N_EDGES);
    // 7) h1 = relu(h1)
    relu_inplace_kernel<<<(N_NODES * HID_DIM + T - 1) / T, T, 0, stream>>>(
        h1, N_NODES * HID_DIM);
    // 8) xw2 = h1 @ W2
    {
        int tiles = mtiles * (OUT_DIM / 16);              // 25000
        gemm_wmma_kernel<OUT_DIM><<<(tiles + 7) / 8, 256, 0, stream>>>(h1, W2, xw2, mtiles);
    }
    // 9) out = dis^2*xw2 + b2
    self_bias_kernel<<<(N_NODES * OUT_DIM + T - 1) / T, T, 0, stream>>>(
        xw2, dis, b2, out, N_NODES * OUT_DIM, OUT_DIM);
    // 10) out[dst] += norm * xw2[src]
    scatter_kernel<OUT_DIM / 32><<<(N_EDGES * 32 + T - 1) / T, T, 0, stream>>>(
        srcp, dstp, ew, dis, xw2, out, N_EDGES);
}
